// GatedGraphAttConvEncoder_76141180223658
// MI455X (gfx1250) — matrix operand order repaired and checked by reading, not compiled, not run on hardware
//
#include <hip/hip_runtime.h>
#include <hip/hip_bf16.h>

typedef __bf16 bf16_t;
typedef __attribute__((ext_vector_type(16))) __bf16 v16bf;
typedef __attribute__((ext_vector_type(8)))  __bf16 v8bf;
typedef __attribute__((ext_vector_type(8)))  float   v8f;

#define B_   2
#define N_   512
#define D_   128
#define H_   8
#define FF_  512
#define DK_  16
#define BN_  (B_*N_)                    /* 1024 node rows   */
#define BNN_ ((size_t)B_*N_*N_)         /* 524288 edges     */
#define EPS_ 1e-5f

__device__ __forceinline__ float gelu_f(float x){
  float x3 = x*x*x;
  return 0.5f*x*(1.0f + tanhf(0.7978845608028654f*(x + 0.044715f*x3)));
}

// ---------------------------------------------------------------------------
// Weight packing: src f32 (generic strides) -> bf16 WMMA-B fragment layout.
// Packed index = ((kt*NTtot + ntBase + nt)*32 + lane)*16 + v
// B-operand lane map (ISA table): K = kt*32 + 8*(lane/16) + (v&7) + (v>=8 ? 16 : 0)
// column n = nt*16 + lane%16.  src index = (k/16)*rS1+(k%16)*rS0+(c/16)*cS1+(c%16)*cS0
// ---------------------------------------------------------------------------
__global__ void pack_weight(const float* __restrict__ src, bf16_t* __restrict__ dst,
                            int K, int Nsrc, int rS1, int rS0, int cS1, int cS0,
                            int NTtot, int ntBase){
  int idx = blockIdx.x*256 + threadIdx.x;
  if (idx >= K*Nsrc) return;
  int v    = idx & 15;
  int lane = (idx >> 4) & 31;
  int t    = idx >> 9;
  int NTs  = Nsrc >> 4;
  int nt   = t % NTs, kt = t / NTs;
  int k    = kt*32 + ((lane >= 16) ? 8 : 0) + (v & 7) + ((v >= 8) ? 16 : 0);
  int c    = nt*16 + (lane & 15);
  float val = src[(k>>4)*rS1 + (k&15)*rS0 + (c>>4)*cS1 + (c&15)*cS0];
  dst[((size_t)(kt*NTtot + ntBase + nt)*32 + lane)*16 + v] = (bf16_t)val;
}

// ---------------------------------------------------------------------------
// Generic row-major-A (bf16) x packed-B GEMM, f32 accumulate via v_wmma.
// Block = 128 thr (4 waves); wave handles one 16-wide N tile; grid (M/16, N/64).
// act: 0=none 1=gelu.  Either/both of outf (f32) / outb (bf16) written.
// ---------------------------------------------------------------------------
__global__ void __launch_bounds__(128) gemm_bf16(
    const bf16_t* __restrict__ A, int lda,
    const bf16_t* __restrict__ Bp, int K, int N,
    const float* __restrict__ bias,
    float* __restrict__ outf, bf16_t* __restrict__ outb, int ldo, int act)
{
  int lane = threadIdx.x & 31;
  int wave = threadIdx.x >> 5;
  int m0   = blockIdx.x << 4;
  int NT   = N >> 4;
  int nt   = (blockIdx.y << 2) + wave;
  if (nt >= NT) return;
  int half = lane >> 4;
  int ml   = m0 + (lane & 15);
  int klo  = half << 3;                 // A-operand lane map (ISA table)
  v8f acc = {};
  int KT = K >> 5;
  for (int kt = 0; kt < KT; ++kt){
    const bf16_t* pa = A + (size_t)ml*lda + (kt<<5) + klo;
    v8bf alo = *(const v8bf*)pa;
    v8bf ahi = *(const v8bf*)(pa + 16);
    v16bf a;
#pragma unroll
    for (int q=0;q<8;q++){ a[q]=alo[q]; a[q+8]=ahi[q]; }
    v16bf b = *(const v16bf*)(Bp + ((size_t)(kt*NT + nt)*32 + lane)*16);
    acc = __builtin_amdgcn_wmma_f32_16x16x32_bf16(false, a, false, b, (short)0,
                                                  acc, false, false);
  }
  int n  = (nt<<4) + (lane & 15);
  float bv = bias ? bias[n] : 0.0f;
  int mb = m0 + (half << 3);
#pragma unroll
  for (int r=0;r<8;r++){
    float v = acc[r] + bv;
    if (act == 1) v = gelu_f(v);
    size_t o = (size_t)(mb + r)*ldo + n;
    if (outf) outf[o] = v;
    if (outb) outb[o] = (bf16_t)v;
  }
}

// ---------------------------------------------------------------------------
// Fused edge pass 1: per (b,i) block computes, for all 512 j:
//   Ce = e[b,i,j,:] @ W_C  (bf16 WMMA, W_C fragments staged in LDS)
//   e_pre = Ce + Ah[i] + Bh[j] + b_C      (optionally stored for pass 2)
//   agg[b,i,:] += sigmoid(e_pre) * !edge * Vh[j]
//   per-channel sum/sumsq of e_pre -> block partials (deterministic reduce)
// node = (BN,512) f32 holding [U|V|A|B]h.
// ---------------------------------------------------------------------------
__global__ void __launch_bounds__(128) edge_pass1(
    const float* __restrict__ e, const bf16_t* __restrict__ WCp,
    const float* __restrict__ node, const float* __restrict__ bC,
    const unsigned char* __restrict__ edges,
    float* __restrict__ epre, float* __restrict__ agg,
    float* __restrict__ pstats, int store_pre)
{
  __shared__ bf16_t sWC[128*128];
  __shared__ float  sRed[4][3][128];
  int tid = threadIdx.x;
  { // stage packed W_C (32 KB) into LDS
    const uint4* s = (const uint4*)WCp;
    uint4* d = (uint4*)sWC;
    for (int i = tid; i < (128*128*2)/16; i += 128) d[i] = s[i];
  }
  __syncthreads();
  int bi = blockIdx.x;              // 0..BN-1  (b,i)
  int b  = bi >> 9;
  int lane = tid & 31, wave = tid >> 5;
  int half = lane >> 4, lidx = lane & 15;
  const float* erow = e    + (size_t)bi*N_*D_;
  float*       prow = epre + (size_t)bi*N_*D_;
  const unsigned char* mrow = edges + (size_t)bi*N_;
  float aggA[8], sumA[8], sqA[8];
#pragma unroll
  for (int t=0;t<8;t++){ aggA[t]=0.f; sumA[t]=0.f; sqA[t]=0.f; }
  v8f zero = {};
  for (int jt = wave; jt < 32; jt += 4){
    int j0 = jt << 4;
    v8f acc[8];
#pragma unroll
    for (int t=0;t<8;t++) acc[t] = zero;
    int m   = j0 + lidx;            // A row (edge j)
    int klo = half << 3;
    for (int kt = 0; kt < 4; ++kt){
      const float* pa = erow + (size_t)m*D_ + (kt<<5) + klo;
      __builtin_prefetch(pa + D_, 0, 1);
      v16bf a;
#pragma unroll
      for (int q=0;q<8;q++){ a[q]=(bf16_t)pa[q]; a[q+8]=(bf16_t)pa[q+16]; }
#pragma unroll
      for (int nt=0;nt<8;++nt){
        v16bf bb = *(const v16bf*)(sWC + ((kt*8 + nt)*32 + lane)*16);
        acc[nt] = __builtin_amdgcn_wmma_f32_16x16x32_bf16(false, a, false, bb,
                                                          (short)0, acc[nt], false, false);
      }
    }
#pragma unroll
    for (int nt=0;nt<8;++nt){
      int d = (nt<<4) + lidx;
      float ah = node[(size_t)bi*512 + 256 + d] + bC[d];
#pragma unroll
      for (int r=0;r<8;++r){
        int j = j0 + r + (half<<3);
        size_t nj = (size_t)(b*N_ + j)*512;
        float pre = acc[nt][r] + ah + node[nj + 384 + d];
        if (store_pre) prow[(size_t)j*D_ + d] = pre;
        float g = 1.0f/(1.0f + __expf(-pre));
        float valid = mrow[j] ? 0.0f : 1.0f;
        aggA[nt] += g * valid * node[nj + 128 + d];
        sumA[nt] += pre;
        sqA[nt]  += pre*pre;
      }
    }
  }
  // deterministic reduce: half-wave shuffle, then fixed-order wave combine
#pragma unroll
  for (int nt=0;nt<8;++nt){
    float a0 = aggA[nt] + __shfl_xor(aggA[nt], 16, 32);
    float s0 = sumA[nt] + __shfl_xor(sumA[nt], 16, 32);
    float q0 = sqA[nt]  + __shfl_xor(sqA[nt],  16, 32);
    if (half == 0){
      int d = (nt<<4) + lidx;
      sRed[wave][0][d] = a0; sRed[wave][1][d] = s0; sRed[wave][2][d] = q0;
    }
  }
  __syncthreads();
  if (tid < 128){
    float a = sRed[0][0][tid]+sRed[1][0][tid]+sRed[2][0][tid]+sRed[3][0][tid];
    float s = sRed[0][1][tid]+sRed[1][1][tid]+sRed[2][1][tid]+sRed[3][1][tid];
    float q = sRed[0][2][tid]+sRed[1][2][tid]+sRed[2][2][tid]+sRed[3][2][tid];
    agg[(size_t)bi*D_ + tid]          = a;
    pstats[(size_t)bi*256 + tid]       = s;
    pstats[(size_t)bi*256 + 128 + tid] = q;
  }
}

__global__ void edge_stats_reduce(const float* __restrict__ pstats, float* __restrict__ st){
  int c = threadIdx.x;                  // 128 channels
  float s=0.f,q=0.f;
  for (int i=0;i<BN_;++i){ s += pstats[(size_t)i*256 + c]; q += pstats[(size_t)i*256 + 128 + c]; }
  st[c]=s; st[128+c]=q;
}

// e += gelu(bn(e_pre))   (elementwise float4 stream)
__global__ void edge_update(const float* __restrict__ epre, const float* __restrict__ g,
                            const float* __restrict__ bta, const float* __restrict__ st,
                            float* __restrict__ e){
  size_t i4 = (size_t)blockIdx.x*256 + threadIdx.x;
  if (i4 >= BNN_*(D_/4)) return;
  int d0 = ((int)(i4 & 31)) << 2;
  const float cnt = (float)BNN_;
  float4 p  = ((const float4*)epre)[i4];
  float4 ev = ((const float4*)e)[i4];
  float pv[4]  = {p.x,p.y,p.z,p.w};
  float evv[4] = {ev.x,ev.y,ev.z,ev.w};
#pragma unroll
  for (int c=0;c<4;++c){
    int d = d0 + c;
    float mean = st[d]/cnt;
    float var  = st[128+d]/cnt - mean*mean;
    float v = (pv[c]-mean)*rsqrtf(var+EPS_)*g[d] + bta[d];
    evv[c] += gelu_f(v);
  }
  ((float4*)e)[i4] = make_float4(evv[0],evv[1],evv[2],evv[3]);
}

// dist BatchNorm1d(1) stats: single block, deterministic tree reduce
__global__ void dist_stats(const float* __restrict__ x, float* __restrict__ st){
  __shared__ float ss[256], sq[256];
  float s=0.f,q=0.f;
  for (size_t i = threadIdx.x; i < BNN_; i += 256){ float v=x[i]; s+=v; q+=v*v; }
  ss[threadIdx.x]=s; sq[threadIdx.x]=q; __syncthreads();
  for (int o=128;o>0;o>>=1){
    if ((int)threadIdx.x < o){ ss[threadIdx.x]+=ss[threadIdx.x+o]; sq[threadIdx.x]+=sq[threadIdx.x+o]; }
    __syncthreads();
  }
  if (threadIdx.x==0){ st[0]=ss[0]; st[1]=sq[0]; }
}

// e = normalize(dist)*We + be   (rank-1 edge-feature init, float4 stream)
__global__ void e_init(const float* __restrict__ dist, const float* __restrict__ dg,
                       const float* __restrict__ db, const float* __restrict__ We,
                       const float* __restrict__ be, const float* __restrict__ st,
                       float* __restrict__ e){
  size_t i4 = (size_t)blockIdx.x*256 + threadIdx.x;
  if (i4 >= BNN_*(D_/4)) return;
  size_t edge = i4 >> 5;
  int d4 = (int)(i4 & 31);
  float mean = st[0]/(float)BNN_;
  float var  = st[1]/(float)BNN_ - mean*mean;
  float dn = (dist[edge]-mean)*rsqrtf(var+EPS_)*dg[0] + db[0];
  float4 w  = ((const float4*)We)[d4];
  float4 bb = ((const float4*)be)[d4];
  ((float4*)e)[i4] = make_float4(dn*w.x+bb.x, dn*w.y+bb.y, dn*w.z+bb.z, dn*w.w+bb.w);
}

// VALU flash attention: block per (b,h,128-q-chunk); K/V (b,h) staged in LDS.
// qkv layout (BN,512) bf16: q[0:128) k[128:256) v[256:384); output -> [384:512)
__global__ void __launch_bounds__(128) attention_k(
    const bf16_t* __restrict__ qkv, const unsigned char* __restrict__ edges,
    bf16_t* __restrict__ outb){
  __shared__ bf16_t sK[N_*DK_];
  __shared__ bf16_t sV[N_*DK_];
  int tid = threadIdx.x;
  int blk = blockIdx.x;                  // B*H*4
  int qc = blk & 3; int bh = blk >> 2; int h = bh & 7; int b = bh >> 3;
  for (int j = tid; j < N_; j += 128){
    const uint4* ks = (const uint4*)(qkv + ((size_t)(b*N_ + j)*512) + 128 + h*16);
    const uint4* vs = (const uint4*)(qkv + ((size_t)(b*N_ + j)*512) + 256 + h*16);
    ((uint4*)(sK + j*16))[0] = ks[0]; ((uint4*)(sK + j*16))[1] = ks[1];
    ((uint4*)(sV + j*16))[0] = vs[0]; ((uint4*)(sV + j*16))[1] = vs[1];
  }
  __syncthreads();
  int qrow = qc*128 + tid;
  const bf16_t* qp = qkv + ((size_t)(b*N_ + qrow)*512) + h*16;
  float qreg[16];
#pragma unroll
  for (int k=0;k<16;k++) qreg[k] = (float)qp[k];
  const unsigned char* er = edges + ((size_t)(b*N_ + qrow))*N_;
  float mx = -1e30f, l = 0.f, o[16];
#pragma unroll
  for (int k=0;k<16;k++) o[k]=0.f;
  for (int j=0;j<N_;++j){
    float s = 0.f;
#pragma unroll
    for (int k=0;k<16;k++) s += qreg[k]*(float)sK[j*16+k];
    s *= 0.25f;                           // 1/sqrt(DK)
    if (er[j]) s = -1e9f;
    float mn = fmaxf(mx, s);
    float sc = __expf(mx - mn);
    float p  = __expf(s - mn);
    l = l*sc + p;
#pragma unroll
    for (int k=0;k<16;k++) o[k] = o[k]*sc + p*(float)sV[j*16+k];
    mx = mn;
  }
  float inv = 1.f/l;
  bf16_t* op = outb + ((size_t)(b*N_ + qrow)*512) + 384 + h*16;
#pragma unroll
  for (int k=0;k<16;k++) op[k] = (bf16_t)(o[k]*inv);
}

// per-channel sum/sumsq over BN rows of (x [+ y]); single block, deterministic
__global__ void node_stats(const float* __restrict__ x, int ldx,
                           const float* __restrict__ y, int ldy,
                           float* __restrict__ st){
  int c = threadIdx.x;
  float s=0.f,q=0.f;
  for (int r=0;r<BN_;++r){
    float v = x[(size_t)r*ldx + c];
    if (y) v += y[(size_t)r*ldy + c];
    s+=v; q+=v*v;
  }
  st[c]=s; st[128+c]=q;
}

// out = [gelu]( bn(x [+ y]) )
__global__ void bn_apply(const float* __restrict__ x, int ldx,
                         const float* __restrict__ y, int ldy,
                         const float* __restrict__ g, const float* __restrict__ bta,
                         const float* __restrict__ st,
                         float* __restrict__ out, int ldo, int act){
  int idx = blockIdx.x*256 + threadIdx.x;
  if (idx >= BN_*D_) return;
  int r = idx >> 7, c = idx & 127;
  const float cnt = (float)BN_;
  float mean = st[c]/cnt;
  float var  = st[128+c]/cnt - mean*mean;
  float rstd = rsqrtf(var + EPS_);
  float v = x[(size_t)r*ldx + c];
  if (y) v += y[(size_t)r*ldy + c];
  v = (v-mean)*rstd*g[c] + bta[c];
  if (act) v = gelu_f(v);
  out[(size_t)r*ldo + c] = v;
}

__global__ void cvt_bf16_k(const float* __restrict__ x, bf16_t* __restrict__ y, int n){
  int i = blockIdx.x*256 + threadIdx.x;
  if (i < n) y[i] = (bf16_t)x[i];
}

// ---------------------------------------------------------------------------
// Orchestration. Input order = pytree flatten (sorted dict keys):
// 0:x 1:dist 2:edges 3:We 4:be 5:dist_b 6:dist_g, then 28 tensors per layer.
// ---------------------------------------------------------------------------
extern "C" void kernel_launch(void* const* d_in, const int* in_sizes, int n_in,
                              void* d_out, int out_size, void* d_ws, size_t ws_size,
                              hipStream_t stream){
  (void)in_sizes; (void)n_in; (void)out_size; (void)ws_size;
  const float* x      = (const float*)d_in[0];
  const float* dist   = (const float*)d_in[1];
  const unsigned char* edges = (const unsigned char*)d_in[2];
  const float* We     = (const float*)d_in[3];
  const float* be     = (const float*)d_in[4];
  const float* dist_b = (const float*)d_in[5];
  const float* dist_g = (const float*)d_in[6];
  enum { iW_A=0,iW_B,iW_C,iW_U,iW_V,iW_f1,iW_f2,iWk,iWo,iWq,iWv,
         ib1,ib2,ib3,ib_A,ib_B,ib_C,ib_U,ib_V,ib_f1,ib_f2,ib_e,ib_h,
         ig1,ig2,ig3,ig_e,ig_h };
  auto LPf = [&](int p, int k)->const float*{ return (const float*)d_in[7 + 28*p + k]; };

  char* w = (char*)d_ws; size_t off = 0;
  auto take = [&](size_t bytes)->char*{
    char* r = w + off; off = (off + bytes + 255) & ~(size_t)255; return r; };
  float*  e     = (float*)take(BNN_*D_*4);           // 256 MiB
  float*  epre  = (float*)take(BNN_*D_*4);           // 256 MiB
  float*  h     = (float*)take((size_t)BN_*D_*4);
  bf16_t* hbf   = (bf16_t*)take((size_t)BN_*D_*2);
  float*  nbuf1 = (float*)take((size_t)BN_*512*4);   // [U|V|A|B]h
  float*  nbuf2 = (float*)take((size_t)BN_*D_*4);
  bf16_t* nbf   = (bf16_t*)take((size_t)BN_*512*2);  // qkv / attn-out / ff1
  float*  agg   = (float*)take((size_t)BN_*D_*4);
  float*  std_  = (float*)take(2*4);
  float*  stn   = (float*)take(256*4);
  float*  ste   = (float*)take(256*4);
  float*  pstats= (float*)take((size_t)BN_*256*4);
  float*  buvab = (float*)take(512*4);
  bf16_t* pw[2][6];
  size_t psz[6] = {128*384, 128*128, 128*512, 128*128, 128*512, 512*128};
  for (int p=0;p<2;p++) for (int q=0;q<6;q++) pw[p][q] = (bf16_t*)take(psz[q]*2);

  // ---- pack all weights into WMMA-B fragment layout (bf16) ----
  for (int p=0;p<2;p++){
    // Wq/Wk/Wv: (H,D,DK) -> rows d, cols h*16+k
    pack_weight<<<64,256,0,stream>>>(LPf(p,iWq), pw[p][0],128,128, 256,16,2048,1, 24,0);
    pack_weight<<<64,256,0,stream>>>(LPf(p,iWk), pw[p][0],128,128, 256,16,2048,1, 24,8);
    pack_weight<<<64,256,0,stream>>>(LPf(p,iWv), pw[p][0],128,128, 256,16,2048,1, 24,16);
    // Wo: (H,DK,D) -> rows h*16+k, cols d
    pack_weight<<<64,256,0,stream>>>(LPf(p,iWo), pw[p][1],128,128, 2048,128,16,1, 8,0);
    // U|V|A|B concat (plain DxD each)
    pack_weight<<<64,256,0,stream>>>(LPf(p,iW_U),pw[p][2],128,128, 2048,128,16,1, 32,0);
    pack_weight<<<64,256,0,stream>>>(LPf(p,iW_V),pw[p][2],128,128, 2048,128,16,1, 32,8);
    pack_weight<<<64,256,0,stream>>>(LPf(p,iW_A),pw[p][2],128,128, 2048,128,16,1, 32,16);
    pack_weight<<<64,256,0,stream>>>(LPf(p,iW_B),pw[p][2],128,128, 2048,128,16,1, 32,24);
    pack_weight<<<64,256,0,stream>>>(LPf(p,iW_C),pw[p][3],128,128, 2048,128,16,1, 8,0);
    pack_weight<<<256,256,0,stream>>>(LPf(p,iW_f1),pw[p][4],128,512, 8192,512,16,1, 32,0);
    pack_weight<<<256,256,0,stream>>>(LPf(p,iW_f2),pw[p][5],512,128, 2048,128,16,1, 8,0);
  }

  // ---- edge feature init ----
  dist_stats<<<1,256,0,stream>>>(dist, std_);
  e_init<<<65536,256,0,stream>>>(dist, dist_g, dist_b, We, be, std_, e);
  hipMemcpyAsync(h, x, (size_t)BN_*D_*4, hipMemcpyDeviceToDevice, stream);

  for (int p=0;p<2;p++){
    // ---- MHA ----
    cvt_bf16_k<<<512,256,0,stream>>>(h, hbf, BN_*D_);
    gemm_bf16<<<dim3(64,6),128,0,stream>>>(hbf,128, pw[p][0],128,384, nullptr,
                                           nullptr, nbf,512, 0);
    attention_k<<<64,128,0,stream>>>(nbf, edges, nbf);
    gemm_bf16<<<dim3(64,2),128,0,stream>>>(nbf+384,512, pw[p][1],128,128, nullptr,
                                           nbuf2, nullptr,128, 0);
    node_stats<<<1,128,0,stream>>>(h,128, nbuf2,128, stn);
    bn_apply<<<512,256,0,stream>>>(h,128, nbuf2,128, LPf(p,ig1),LPf(p,ib1), stn, h,128, 0);

    // ---- GatedGCN ----
    cvt_bf16_k<<<512,256,0,stream>>>(h, hbf, BN_*D_);
    hipMemcpyAsync(buvab+0,   LPf(p,ib_U), 512, hipMemcpyDeviceToDevice, stream);
    hipMemcpyAsync(buvab+128, LPf(p,ib_V), 512, hipMemcpyDeviceToDevice, stream);
    hipMemcpyAsync(buvab+256, LPf(p,ib_A), 512, hipMemcpyDeviceToDevice, stream);
    hipMemcpyAsync(buvab+384, LPf(p,ib_B), 512, hipMemcpyDeviceToDevice, stream);
    gemm_bf16<<<dim3(64,8),128,0,stream>>>(hbf,128, pw[p][2],128,512, buvab,
                                           nbuf1, nullptr,512, 0);
    edge_pass1<<<BN_,128,0,stream>>>(e, pw[p][3], nbuf1, LPf(p,ib_C), edges,
                                     epre, agg, pstats, (p==0)?1:0);
    if (p==0){ // e is dead after last layer: skip e-update there
      edge_stats_reduce<<<1,128,0,stream>>>(pstats, ste);
      edge_update<<<65536,256,0,stream>>>(epre, LPf(p,ig_e), LPf(p,ib_e), ste, e);
    }
    node_stats<<<1,128,0,stream>>>(nbuf1,512, agg,128, stn);
    bn_apply<<<512,256,0,stream>>>(nbuf1,512, agg,128, LPf(p,ig_h),LPf(p,ib_h),
                                   stn, nbuf2,128, 1);
    node_stats<<<1,128,0,stream>>>(h,128, nbuf2,128, stn);
    bn_apply<<<512,256,0,stream>>>(h,128, nbuf2,128, LPf(p,ig2),LPf(p,ib2), stn, h,128, 0);

    // ---- FFN ----
    cvt_bf16_k<<<512,256,0,stream>>>(h, hbf, BN_*D_);
    gemm_bf16<<<dim3(64,8),128,0,stream>>>(hbf,128, pw[p][4],128,512, LPf(p,ib_f1),
                                           nullptr, nbf,512, 1);
    gemm_bf16<<<dim3(64,2),128,0,stream>>>(nbf,512, pw[p][5],512,128, LPf(p,ib_f2),
                                           nbuf2, nullptr,128, 0);
    node_stats<<<1,128,0,stream>>>(h,128, nbuf2,128, stn);
    float* outp = (p==1) ? (float*)d_out : h;
    bn_apply<<<512,256,0,stream>>>(h,128, nbuf2,128, LPf(p,ig3),LPf(p,ib3), stn, outp,128, 0);
  }
}
